// Routing_10136122818743
// MI455X (gfx1250) — matrix operand order repaired and checked by reading, not compiled, not run on hardware
//
#include <hip/hip_runtime.h>
#include <hip/hip_bf16.h>

typedef __attribute__((ext_vector_type(16))) _Float16 v16h;
typedef __attribute__((ext_vector_type(8)))  float    v8f;
typedef __attribute__((ext_vector_type(4)))  float    f32x4;

#define B_    16
#define WORDS 32
#define DIM   256
#define DH    32      // DIM_HEAD
#define INNER 256
#define J_    256     // WORDS*HEADS
#define N_    8192    // WORDS*INNER (flattened i,e)
#define JC    32      // j-chunks for routing (8 rows per chunk)

// ---------------------------------------------------------------------------
// wave32 butterfly reductions
// ---------------------------------------------------------------------------
__device__ __forceinline__ float wave_sum(float v) {
#pragma unroll
  for (int off = 16; off > 0; off >>= 1) v += __shfl_xor(v, off, 32);
  return v;
}
__device__ __forceinline__ float wave_max(float v) {
#pragma unroll
  for (int off = 16; off > 0; off >>= 1) v = fmaxf(v, __shfl_xor(v, off, 32));
  return v;
}

// ---------------------------------------------------------------------------
// Kernel 1: xi = x @ Wi.T + bi, reshape (b,n,(h d)) -> (b, n*8+h, d), cvt f16
// ---------------------------------------------------------------------------
__global__ void inner_proj_kernel(const float* __restrict__ x,
                                  const float* __restrict__ Wi,
                                  const float* __restrict__ bi,
                                  _Float16* __restrict__ xh) {
  __shared__ float xs[DIM];
  const int row = blockIdx.x;            // b*32 + n
  const int m   = threadIdx.x;           // inner index 0..255
  xs[m] = x[(size_t)row * DIM + m];
  __syncthreads();
  float acc = bi[m];
  const float* wr = Wi + (size_t)m * DIM;
#pragma unroll 8
  for (int k = 0; k < DIM; ++k) acc = fmaf(xs[k], wr[k], acc);
  const int b = row >> 5, n = row & 31;
  const int j = n * 8 + (m >> 5);
  const int d = m & 31;
  xh[((size_t)b * J_ + j) * DH + d] = (_Float16)acc;
}

// ---------------------------------------------------------------------------
// Kernel 2: u_hat[b,j,n] = sum_d W[j,n,d]*xh[b,j,d] + bias[j,n]
// Per j: C[16 x 8192] = A[16x32] * B[32x8192]; K = DH = 32 -> exactly one
// v_wmma_f32_16x16x32_f16 per 16x16 C tile. W is streamed once (256 MiB) ->
// non-temporal loads so u_hat (134 MB, re-read 5x by routing) stays in L2.
// ---------------------------------------------------------------------------
__global__ void uhat_wmma_kernel(const float* __restrict__ W,
                                 const float* __restrict__ bias,
                                 const _Float16* __restrict__ xh,
                                 float* __restrict__ uhat) {
  const int j    = blockIdx.y;
  const int lane = threadIdx.x & 31;
  const int wave = threadIdx.x >> 5;
  const int waveInJ = blockIdx.x * 8 + wave;   // 0..63
  const int nl = lane & 15;
  const bool hi = lane >= 16;

  // A: xh[0..15, j, 0..31], row m = lane%16, k = 16*(lane/16) + s (contiguous, same as B)
  v16h a;
  {
    const _Float16* arow = xh + ((size_t)nl * J_ + j) * DH;
    const int k0 = hi ? 16 : 0;
    const int k1 = hi ? 24 : 8;
#pragma unroll
    for (int p = 0; p < 8; ++p) a[p] = arow[k0 + p];
#pragma unroll
    for (int p = 0; p < 8; ++p) a[8 + p] = arow[k1 + p];
  }

#pragma unroll 2
  for (int t = 0; t < 8; ++t) {
    const int ntile = waveInJ * 8 + t;       // 0..511
    const int n = ntile * 16 + nl;           // flattened (i,e) column
    // B column n: W[j, n, d] contiguous in d; stream with NT hint, cvt -> f16
    const float* wrow = W + ((size_t)j * N_ + n) * DH + (hi ? 16 : 0);
    v16h bm;
#pragma unroll
    for (int p = 0; p < 16; p += 4) {
      f32x4 wv = __builtin_nontemporal_load((const f32x4*)(wrow + p));
      bm[p + 0] = (_Float16)wv.x;
      bm[p + 1] = (_Float16)wv.y;
      bm[p + 2] = (_Float16)wv.z;
      bm[p + 3] = (_Float16)wv.w;
    }
    v8f c = {};
    c = __builtin_amdgcn_wmma_f32_16x16x32_f16(
        /*neg_a=*/false, a, /*neg_b=*/false, bm,
        /*c_mod=*/(short)0, c, /*reuse_a=*/false, /*reuse_b=*/false);
    const float bv = bias[(size_t)j * N_ + n];
    float* ob = uhat + (size_t)j * N_ + n;
#pragma unroll
    for (int r = 0; r < 8; ++r) {
      const int brow = r + (hi ? 8 : 0);     // batch index = M
      ob[(size_t)brow * J_ * N_] = c[r] + bv;
    }
  }
}

// ---------------------------------------------------------------------------
// Kernel 3a: one routing pass. grid (JC=32 chunks, B=16), block 256 (8 waves,
// one j row per wave). Per row: (it>0) logits += <u,v>, softmax over i,
// then partial s accumulation in LDS; partial tile written to scratch.
// logits rows are wave-private across launches -> race-free, deterministic.
// ---------------------------------------------------------------------------
__global__ void routing_pass_kernel(const float* __restrict__ uhat,
                                    float* __restrict__ logits,
                                    const float* __restrict__ vin,
                                    float* __restrict__ spart,
                                    int it) {
  __shared__ float vsh[WORDS * INNER];    // 32 KB
  __shared__ float sacc[WORDS * INNER];   // 32 KB
  const int chunk = blockIdx.x;
  const int b     = blockIdx.y;
  const int tid   = threadIdx.x;
  const int lane  = tid & 31;
  const int wave  = tid >> 5;
  const int j     = chunk * 8 + wave;

  if (it > 0)
    for (int idx = tid; idx < WORDS * INNER; idx += 256)
      vsh[idx] = vin[(size_t)b * WORDS * INNER + idx];
  for (int idx = tid; idx < WORDS * INNER; idx += 256) sacc[idx] = 0.f;
  __syncthreads();

  const float* uj = uhat + ((size_t)b * J_ + j) * N_;
  float* lg = logits + ((size_t)b * J_ + j) * WORDS;
  float c_l;                               // coupling coeff for i = lane
  if (it > 0) {
    float mylg = lg[lane];
#pragma unroll 1
    for (int i = 0; i < WORDS; ++i) {
      const float* ui = uj + i * INNER;
      float d = 0.f;
#pragma unroll
      for (int q = 0; q < 8; ++q) {
        const int e = q * 32 + lane;
        d = fmaf(ui[e], vsh[i * INNER + e], d);
      }
      d = wave_sum(d);
      mylg += (lane == i) ? d : 0.f;
    }
    lg[lane] = mylg;
    const float mx = wave_max(mylg);       // softmax over i (= lanes)
    const float ex = __expf(mylg - mx);
    const float sm = wave_sum(ex);
    c_l = ex / sm;
  } else {
    lg[lane] = 0.f;                        // init for iteration 1
    c_l = 1.0f / 32.0f;                    // b_log == 0 -> uniform
  }
#pragma unroll 1
  for (int i = 0; i < WORDS; ++i) {
    const float cv = __shfl(c_l, i, 32);
    const float* ui = uj + i * INNER;
#pragma unroll
    for (int q = 0; q < 8; ++q) {
      const int e = q * 32 + lane;
      atomicAdd(&sacc[i * INNER + e], cv * ui[e]);
    }
  }
  __syncthreads();
  float* sp = spart + ((size_t)b * JC + chunk) * (WORDS * INNER);
  for (int idx = tid; idx < WORDS * INNER; idx += 256) sp[idx] = sacc[idx];
}

// ---------------------------------------------------------------------------
// Kernel 3b: reduce partial s over chunks + squash. grid (i=32, b=16),
// block 256 (e = tid). v = s * n/(1+n^2).
// ---------------------------------------------------------------------------
__global__ void squash_kernel(const float* __restrict__ spart,
                              float* __restrict__ vout) {
  __shared__ float wred[8];
  const int i   = blockIdx.x;
  const int b   = blockIdx.y;
  const int e   = threadIdx.x;
  const int lane = e & 31, wave = e >> 5;
  float s = 0.f;
#pragma unroll
  for (int c = 0; c < JC; ++c)
    s += spart[((size_t)b * JC + c) * (WORDS * INNER) + i * INNER + e];
  float p = wave_sum(s * s);
  if (lane == 0) wred[wave] = p;
  __syncthreads();
  float nsq = 0.f;
#pragma unroll
  for (int w = 0; w < 8; ++w) nsq += wred[w];
  const float nrm = sqrtf(nsq);
  const float f = nrm / (1.f + nsq);
  vout[(size_t)b * WORDS * INNER + i * INNER + e] = s * f;
}

// ---------------------------------------------------------------------------
// Kernel 4: out = v @ Wo.T + bo
// ---------------------------------------------------------------------------
__global__ void out_proj_kernel(const float* __restrict__ v,
                                const float* __restrict__ Wo,
                                const float* __restrict__ bo,
                                float* __restrict__ out) {
  __shared__ float vs[INNER];
  const int row = blockIdx.x;              // b*32 + w
  const int m   = threadIdx.x;
  vs[m] = v[(size_t)row * INNER + m];
  __syncthreads();
  float acc = bo[m];
  const float* wr = Wo + (size_t)m * INNER;
#pragma unroll 8
  for (int k = 0; k < INNER; ++k) acc = fmaf(vs[k], wr[k], acc);
  out[(size_t)row * DIM + m] = acc;
}

// ---------------------------------------------------------------------------
extern "C" void kernel_launch(void* const* d_in, const int* in_sizes, int n_in,
                              void* d_out, int out_size, void* d_ws, size_t ws_size,
                              hipStream_t stream) {
  const float* x    = (const float*)d_in[0];
  const float* W    = (const float*)d_in[1];
  const float* bias = (const float*)d_in[2];
  const float* Wi   = (const float*)d_in[3];
  const float* bi   = (const float*)d_in[4];
  const float* Wo   = (const float*)d_in[5];
  const float* bo   = (const float*)d_in[6];
  float* out = (float*)d_out;

  char* ws = (char*)d_ws;
  const size_t uhat_bytes = (size_t)B_ * J_ * N_ * sizeof(float);      // 134 MB
  const size_t xh_bytes   = (size_t)B_ * J_ * DH * sizeof(_Float16);   // 256 KB
  const size_t v_bytes    = (size_t)B_ * WORDS * INNER * sizeof(float);// 512 KB
  const size_t lg_bytes   = (size_t)B_ * J_ * WORDS * sizeof(float);   // 512 KB
  float*    uhat   = (float*)ws;
  _Float16* xh     = (_Float16*)(ws + uhat_bytes);
  float*    vbuf   = (float*)(ws + uhat_bytes + xh_bytes);
  float*    logits = (float*)(ws + uhat_bytes + xh_bytes + v_bytes);
  float*    spart  = (float*)(ws + uhat_bytes + xh_bytes + v_bytes + lg_bytes); // 16 MB

  inner_proj_kernel<<<dim3(512),     dim3(256), 0, stream>>>(x, Wi, bi, xh);
  uhat_wmma_kernel <<<dim3(8, 256),  dim3(256), 0, stream>>>(W, bias, xh, uhat);
  for (int it = 0; it < 3; ++it) {
    routing_pass_kernel<<<dim3(JC, B_), dim3(256), 0, stream>>>(uhat, logits, vbuf, spart, it);
    squash_kernel      <<<dim3(WORDS, B_), dim3(256), 0, stream>>>(spart, vbuf);
  }
  out_proj_kernel<<<dim3(512), dim3(256), 0, stream>>>(vbuf, Wo, bo, out);
}
